// LoraLinear_66022237274785
// MI455X (gfx1250) — matrix operand order repaired and checked by reading, not compiled, not run on hardware
//
#include <hip/hip_runtime.h>
#include <math.h>

// ---------------- types ----------------
typedef __attribute__((ext_vector_type(16))) __bf16 v16bf;
typedef __attribute__((ext_vector_type(8)))  __bf16 v8bf;
typedef __attribute__((ext_vector_type(4)))  __bf16 v4bf;
typedef __attribute__((ext_vector_type(8)))  float  v8f;
typedef int v4i_async __attribute__((vector_size(16)));   // matches builtin param: int __vector(4)

#define D_MODEL 4096
#define M_ROWS  8192            // B*S = 4*2048
#define K_FOLD  128             // N_LORA * R_LORA
#define N_STAGES 128            // 4096 / 32

// workspace byte offsets
#define OFF_XH   0ull
#define OFF_XL   67108864ull    // 8192*4096*2
#define OFF_WH   134217728ull
#define OFF_WL   167772160ull   // + 4096*4096*2
#define OFF_BS   201326592ull
#define OFF_AFT  202375168ull   // + 4096*128*2

// ---- CDNA5 async global->LDS copy (16B per lane), with safe fallback ----
#if defined(__HIP_DEVICE_COMPILE__) && __has_builtin(__builtin_amdgcn_global_load_async_to_lds_b128)
#define ASYNC_COPY16(gsrc, ldst)                                              \
  __builtin_amdgcn_global_load_async_to_lds_b128(                             \
      (v4i_async*)(gsrc), (v4i_async*)(ldst), 0, 0)
#if __has_builtin(__builtin_amdgcn_s_wait_asynccnt)
#define WAIT_ASYNC(n) __builtin_amdgcn_s_wait_asynccnt(n)
#else
#define WAIT_ASYNC(n) asm volatile("s_wait_asynccnt %0" ::"n"(n))
#endif
#else
// fallback: plain 16B global load + LDS store; barrier provides ordering
#define ASYNC_COPY16(gsrc, ldst) (*(v8bf*)(ldst) = *(const v8bf*)(gsrc))
#define WAIT_ASYNC(n) ((void)0)
#endif

static __device__ __forceinline__ v8f wmma_bf16(v16bf a, v16bf b, v8f c) {
  // (neg_a, A, neg_b, B, c_mod, C, reuse_a, reuse_b)
  return __builtin_amdgcn_wmma_f32_16x16x32_bf16(false, a, false, b, (short)0, c,
                                                 false, false);
}

static __device__ __forceinline__ __bf16 bf16_hi(float x) {
  return (__bf16)x;
}
static __device__ __forceinline__ __bf16 bf16_lo(float x, __bf16 h) {
  return (__bf16)(x - (float)h);
}

// ---------------- kernel 1: split x into bf16 hi/lo ----------------
__global__ __launch_bounds__(256)
void k_convert_x(const float* __restrict__ x,
                 unsigned short* __restrict__ xh_, unsigned short* __restrict__ xl_) {
  __bf16* xh = (__bf16*)xh_;
  __bf16* xl = (__bf16*)xl_;
  unsigned i = blockIdx.x * 256u + threadIdx.x;          // one float4 each
  const float4 v = ((const float4*)x)[i];
  v4bf h, l;
  __bf16 h0 = bf16_hi(v.x); h[0] = h0; l[0] = bf16_lo(v.x, h0);
  __bf16 h1 = bf16_hi(v.y); h[1] = h1; l[1] = bf16_lo(v.y, h1);
  __bf16 h2 = bf16_hi(v.z); h[2] = h2; l[2] = bf16_lo(v.z, h2);
  __bf16 h3 = bf16_hi(v.w); h[3] = h3; l[3] = bf16_lo(v.w, h3);
  ((v4bf*)xh)[i] = h;
  ((v4bf*)xl)[i] = l;
}

// ---------------- kernel 2: softmax + fold lora into [o][128] / [i][128] ----------------
__global__ __launch_bounds__(256)
void k_fold(const float* __restrict__ score,
            const float* __restrict__ lora_A,   // [8][16][4096]
            const float* __restrict__ lora_B,   // [8][4096][16]
            unsigned short* __restrict__ Bs_,   // [4096][128]  s_n * B
            unsigned short* __restrict__ Aft_)  // [4096][128]  A transposed
{
  __bf16* Bs  = (__bf16*)Bs_;
  __bf16* Aft = (__bf16*)Aft_;
  unsigned idx = blockIdx.x * 256u + threadIdx.x;        // 1,048,576 total
  if (idx < 524288u) {
    float mx = score[0];
    #pragma unroll
    for (int n = 1; n < 8; ++n) mx = fmaxf(mx, score[n]);
    float e[8], sum = 0.f;
    #pragma unroll
    for (int n = 0; n < 8; ++n) { e[n] = __expf(score[n] - mx); sum += e[n]; }
    unsigned o = idx >> 7, k = idx & 127u;
    unsigned n = k >> 4, r = k & 15u;
    float s = e[n] / sum;
    Bs[o * 128u + k] = (__bf16)(s * lora_B[(n * 4096u + o) * 16u + r]);
  } else {
    unsigned j = idx - 524288u;
    unsigned i = j >> 7, k = j & 127u;                   // k = n*16 + r
    Aft[i * 128u + k] = (__bf16)lora_A[k * 4096u + i];
  }
}

// ---------------- kernel 3: W_adj = W + Bs @ Aft^T, split to bf16 hi/lo ----------------
__global__ __launch_bounds__(256)
void k_build_w(const float* __restrict__ baseW,
               const unsigned short* __restrict__ Bs_,
               const unsigned short* __restrict__ Aft_,
               unsigned short* __restrict__ Wh_, unsigned short* __restrict__ Wl_) {
  const __bf16* Bs  = (const __bf16*)Bs_;
  const __bf16* Aft = (const __bf16*)Aft_;
  __bf16* Wh = (__bf16*)Wh_;
  __bf16* Wl = (__bf16*)Wl_;

  const int lane = threadIdx.x & 31;
  const int wid  = threadIdx.x >> 5;
  const int tidx = blockIdx.x * 8 + wid;                 // 65536 16x16 tiles
  const int to = (tidx >> 8) << 4;                       // o tile base (M)
  const int ti = (tidx & 255) << 4;                      // i tile base (N)
  const int l15  = lane & 15;
  const int kb8  = (lane >> 4) << 3;                     // 0 / 8
  const int kb16 = (lane >> 4) << 4;                     // 0 / 16

  const __bf16* arow = Bs  + (to + l15) * K_FOLD;        // A-frag: fixed row per lane
  const __bf16* brow = Aft + (ti + l15) * K_FOLD;        // B-frag: fixed col per lane

  v8f acc = {0.f, 0.f, 0.f, 0.f, 0.f, 0.f, 0.f, 0.f};
  #pragma unroll
  for (int ks = 0; ks < K_FOLD; ks += 32) {
    union { v16bf v; v8bf h[2]; } ua;
    ua.h[0] = *(const v8bf*)(arow + ks + kb8);           // K = ks + kb8 .. +7
    ua.h[1] = *(const v8bf*)(arow + ks + 16 + kb8);      // K = ks+16+kb8 .. +7
    union { v16bf v; v8bf h[2]; } ub;
    ub.h[0] = *(const v8bf*)(brow + ks + kb8);           // K = ks + kb8 .. +7
    ub.h[1] = *(const v8bf*)(brow + ks + 16 + kb8);      // K = ks+16+kb8 .. +7
    v16bf b = ub.v;
    acc = wmma_bf16(ua.v, b, acc);
  }
  #pragma unroll
  for (int d = 0; d < 8; ++d) {
    int o = to + d + kb8;                                // C layout: M = d (+8 for hi lanes)
    int i = ti + l15;
    float w = baseW[o * D_MODEL + i] + acc[d];
    __bf16 h = bf16_hi(w);
    __bf16 l = bf16_lo(w, h);
    Wh[o * D_MODEL + i] = h;
    Wl[o * D_MODEL + i] = l;
  }
}

// ---------------- kernel 4: main GEMM out = x @ W_adj^T + bias ----------------
// bf16x3 split, fp32 accum, double-buffered async LDS staging (6 B128 per thread/stage)
__global__ __launch_bounds__(256)
void k_gemm(const unsigned short* __restrict__ xh_, const unsigned short* __restrict__ xl_,
            const unsigned short* __restrict__ Wh_, const unsigned short* __restrict__ Wl_,
            const float* __restrict__ bias, float* __restrict__ out) {
  const __bf16* xh = (const __bf16*)xh_;
  const __bf16* xl = (const __bf16*)xl_;
  const __bf16* Wh = (const __bf16*)Wh_;
  const __bf16* Wl = (const __bf16*)Wl_;

  __shared__ alignas(64) __bf16 sXh[2][128][32];   // 16 KB
  __shared__ alignas(64) __bf16 sXl[2][128][32];   // 16 KB
  __shared__ alignas(64) __bf16 sWh[2][64][32];    //  8 KB
  __shared__ alignas(64) __bf16 sWl[2][64][32];    //  8 KB

  const int t    = threadIdx.x;
  const int lane = t & 31;
  const int wid  = t >> 5;                      // 8 waves
  const int wm   = wid & 3;                     // 4-way M split (32 rows each)
  const int wn   = wid >> 2;                    // 2-way N split (32 cols each)
  const int m0   = blockIdx.y * 128;
  const int n0   = blockIdx.x * 64;
  const int l15  = lane & 15;
  const int kb8  = (lane >> 4) << 3;
  const int kb16 = (lane >> 4) << 4;

  // ---- copy mapping: 16B chunks; X tiles: 512 chunks -> 2/thread; W: 256 -> 1/thread each
  const int cx0 = t * 2, cx1 = t * 2 + 1;
  const int xr0 = cx0 >> 2, xc0 = (cx0 & 3) * 8;   // 128 rows x 4 chunks
  const int xr1 = cx1 >> 2, xc1 = (cx1 & 3) * 8;
  const int wr  = t >> 2,  wc  = (t & 3) * 8;      // 64 rows x 4 chunks

  const __bf16* gxh0 = xh + (m0 + xr0) * D_MODEL + xc0;
  const __bf16* gxh1 = xh + (m0 + xr1) * D_MODEL + xc1;
  const __bf16* gxl0 = xl + (m0 + xr0) * D_MODEL + xc0;
  const __bf16* gxl1 = xl + (m0 + xr1) * D_MODEL + xc1;
  const __bf16* gwh  = Wh + (n0 + wr) * D_MODEL + wc;
  const __bf16* gwl  = Wl + (n0 + wr) * D_MODEL + wc;

  v8f z = {0.f, 0.f, 0.f, 0.f, 0.f, 0.f, 0.f, 0.f};
  v8f acc[2][2];
  acc[0][0] = z; acc[0][1] = z; acc[1][0] = z; acc[1][1] = z;

  // prologue: stage 0 into buffer 0
  {
    ASYNC_COPY16(gxh0, &sXh[0][xr0][xc0]);
    ASYNC_COPY16(gxh1, &sXh[0][xr1][xc1]);
    ASYNC_COPY16(gxl0, &sXl[0][xr0][xc0]);
    ASYNC_COPY16(gxl1, &sXl[0][xr1][xc1]);
    ASYNC_COPY16(gwh,  &sWh[0][wr][wc]);
    ASYNC_COPY16(gwl,  &sWl[0][wr][wc]);
  }

  for (int i = 0; i < N_STAGES; ++i) {
    const int cur = i & 1;
    const int nxt = cur ^ 1;
    const int koff = (i + 1) * 32;                // element offset of next stage
    if (i + 1 < N_STAGES) {
      ASYNC_COPY16(gxh0 + koff, &sXh[nxt][xr0][xc0]);
      ASYNC_COPY16(gxh1 + koff, &sXh[nxt][xr1][xc1]);
      ASYNC_COPY16(gxl0 + koff, &sXl[nxt][xr0][xc0]);
      ASYNC_COPY16(gxl1 + koff, &sXl[nxt][xr1][xc1]);
      ASYNC_COPY16(gwh  + koff, &sWh[nxt][wr][wc]);
      ASYNC_COPY16(gwl  + koff, &sWl[nxt][wr][wc]);
      WAIT_ASYNC(6);                              // retire stage i (in-order), keep i+1 in flight
    } else {
      WAIT_ASYNC(0);
    }
    __syncthreads();                              // publish stage i LDS to all waves

    v16bf ah[2], al[2], bh[2], bl[2];
    #pragma unroll
    for (int mt = 0; mt < 2; ++mt) {
      const __bf16* rh = &sXh[cur][wm * 32 + mt * 16 + l15][0];
      const __bf16* rl = &sXl[cur][wm * 32 + mt * 16 + l15][0];
      union { v16bf v; v8bf h[2]; } u;
      u.h[0] = *(const v8bf*)(rh + kb8);
      u.h[1] = *(const v8bf*)(rh + 16 + kb8);
      ah[mt] = u.v;
      u.h[0] = *(const v8bf*)(rl + kb8);
      u.h[1] = *(const v8bf*)(rl + 16 + kb8);
      al[mt] = u.v;
    }
    #pragma unroll
    for (int nt = 0; nt < 2; ++nt) {
      union { v16bf v; v8bf h[2]; } ub;
      ub.h[0] = *(const v8bf*)&sWh[cur][wn * 32 + nt * 16 + l15][kb8];
      ub.h[1] = *(const v8bf*)&sWh[cur][wn * 32 + nt * 16 + l15][16 + kb8];
      bh[nt] = ub.v;
      ub.h[0] = *(const v8bf*)&sWl[cur][wn * 32 + nt * 16 + l15][kb8];
      ub.h[1] = *(const v8bf*)&sWl[cur][wn * 32 + nt * 16 + l15][16 + kb8];
      bl[nt] = ub.v;
    }
    #pragma unroll
    for (int mt = 0; mt < 2; ++mt)
      #pragma unroll
      for (int nt = 0; nt < 2; ++nt) {
        acc[mt][nt] = wmma_bf16(ah[mt], bh[nt], acc[mt][nt]);  // hi*hi
        acc[mt][nt] = wmma_bf16(ah[mt], bl[nt], acc[mt][nt]);  // hi*lo
        acc[mt][nt] = wmma_bf16(al[mt], bh[nt], acc[mt][nt]);  // lo*hi
      }
    __syncthreads();                              // all reads of buf[cur] done before stage i+2 overwrites
  }

  // epilogue: + bias, fp32 store (coalesced: 16 lanes per row)
  #pragma unroll
  for (int nt = 0; nt < 2; ++nt) {
    const int col = n0 + wn * 32 + nt * 16 + l15;
    const float bv = bias[col];
    #pragma unroll
    for (int mt = 0; mt < 2; ++mt) {
      #pragma unroll
      for (int d = 0; d < 8; ++d) {
        const int row = m0 + wm * 32 + mt * 16 + d + kb8;
        out[row * D_MODEL + col] = acc[mt][nt][d] + bv;
      }
    }
  }
}

// ---------------- launcher ----------------
extern "C" void kernel_launch(void* const* d_in, const int* in_sizes, int n_in,
                              void* d_out, int out_size, void* d_ws, size_t ws_size,
                              hipStream_t stream) {
  const float* x     = (const float*)d_in[0];   // [4,2048,4096]
  const float* baseW = (const float*)d_in[1];   // [4096,4096]
  const float* bias  = (const float*)d_in[2];   // [4096]
  const float* score = (const float*)d_in[3];   // [8]
  const float* lA    = (const float*)d_in[4];   // [8,16,4096]
  const float* lB    = (const float*)d_in[5];   // [8,4096,16]
  float* out = (float*)d_out;

  char* ws = (char*)d_ws;
  unsigned short* xh  = (unsigned short*)(ws + OFF_XH);
  unsigned short* xl  = (unsigned short*)(ws + OFF_XL);
  unsigned short* Wh  = (unsigned short*)(ws + OFF_WH);
  unsigned short* Wl  = (unsigned short*)(ws + OFF_WL);
  unsigned short* Bs  = (unsigned short*)(ws + OFF_BS);
  unsigned short* Aft = (unsigned short*)(ws + OFF_AFT);

  // 1) x -> bf16 hi/lo  (33,554,432 elems / 4 per thread)
  k_convert_x<<<32768, 256, 0, stream>>>(x, xh, xl);
  // 2) softmax + fold lora operands
  k_fold<<<4096, 256, 0, stream>>>(score, lA, lB, Bs, Aft);
  // 3) W_adj = W + delta (rank-128 WMMA GEMM), split hi/lo
  k_build_w<<<8192, 256, 0, stream>>>(baseW, Bs, Aft, Wh, Wl);
  // 4) main GEMM: 64 x 64 blocks of 128x64
  dim3 grid(64, 64);
  k_gemm<<<grid, 256, 0, stream>>>(xh, xl, Wh, Wl, bias, out);
}
